// IVA_36859409334754
// MI455X (gfx1250) — matrix operand order repaired and checked by reading, not compiled, not run on hardware
//
#include <hip/hip_runtime.h>
#include <hip/hip_bf16.h>
#include <math.h>

// ---------- CDNA5 WMMA types ----------
typedef __attribute__((ext_vector_type(16))) _Float16 v16h;
typedef __attribute__((ext_vector_type(8)))  float    v8f;

// Problem constants (match reference setup_inputs)
#define BN   2
#define TN   512
#define FN   1024
#define SG   4          // NUM_GROUPS
#define FQ   256        // F / NUM_GROUPS
#define HC   16         // hidden channels
#define NITER 4
#define FT   (FQ * TN)  // per (b,s) image size

// conv kernel tiling
#define TC   64         // t-chunk
#define H1W  80         // staged h1 width (t local, 5 WMMA tiles of 16)
#define H1R  18         // f rows incl +-1 halo
#define XW   84         // staged x width ([tc-4, tc+80))
#define XR   20         // staged x rows ([f0-2, f0+18))

// ---------------- reshape x (B,T,F) -> Y (B,S,f,T) ----------------
__global__ __launch_bounds__(256) void reshape_in(const float* __restrict__ x,
                                                  float* __restrict__ Y) {
    int idx = blockIdx.x * 256 + threadIdx.x;       // over B*S*FQ*TN = 1M
    int img = idx / FT;
    int rem = idx % FT;
    int f = rem / TN, t = rem % TN;
    int b = img / SG, s = img % SG;
    Y[idx] = x[((size_t)b * TN + t) * FN + s * FQ + f];
}

// ---------------- init W and A to identity over (s,c) ----------------
__global__ __launch_bounds__(256) void init_WA(float* __restrict__ W,
                                               float* __restrict__ A) {
    int idx = blockIdx.x * 256 + threadIdx.x;       // over B*FQ*TN*16
    int sc = idx & 15;
    float v = ((sc >> 2) == (sc & 3)) ? 1.0f : 0.0f;
    W[idx] = v;
    A[idx] = v;
}

// ---------------- mask-net convs: WMMA conv1 + WMMA conv2 -> logits ----------------
__global__ __launch_bounds__(256) void conv_kernel(const float* __restrict__ Y,
                                                   const float* __restrict__ w1,
                                                   const float* __restrict__ b1,
                                                   const float* __restrict__ w2,
                                                   const float* __restrict__ b2,
                                                   float* __restrict__ h2) {
    __shared__ float    xsf[XR * XW];        // staged input tile (f32, zero-padded)
    __shared__ _Float16 h1[HC][H1R][H1W];    // conv1 activations (f16)
    __shared__ _Float16 w1s[HC][9];
    __shared__ _Float16 w2h[160];            // w2 flattened K = c*9+tap, padded to 160
    __shared__ float    b1s[HC];

    int img = blockIdx.x >> 4;       // 8 images
    int ft  = blockIdx.x & 15;       // 16 f-tiles of 16 rows
    int f0  = ft * 16;
    int tid = threadIdx.x;
    int lane = tid & 31, wv = tid >> 5;

    if (tid < HC * 9) w1s[tid / 9][tid % 9] = (_Float16)w1[tid];
    if (tid < 160)    w2h[tid] = (tid < 144) ? (_Float16)w2[tid] : (_Float16)0.0f;
    if (tid < HC)     b1s[tid] = b1[tid];
    float b2v = b2[0];
    __syncthreads();

    // --- WMMA A operand for conv1 (16x32 f16; taps 0..8 in K=0..8, rest zero) ---
    // Contiguous K order: lanes 0-15 row M=lane hold K0..15 in elems 0..15;
    // lanes 16-31 row M=lane-16 hold K16..31 in elems 0..15.
    v16h amat;
    {
        int c = lane & 15;
        for (int e = 0; e < 16; ++e) amat[e] = (_Float16)0.0f;
        if (lane < 16) {
            for (int e = 0; e < 9; ++e) amat[e] = w1s[c][e];   // taps 0..8
        }
    }
    // --- WMMA A operands for conv2: 5 K-chunks of 32, every row = w2 vector ---
    v16h amat2[5];
    for (int j = 0; j < 5; ++j) {
        for (int e = 0; e < 16; ++e) {
            int khalf = (e < 8) ? e : (e - 8 + 16);
            int K = 32 * j + ((lane >= 16) ? 8 : 0) + khalf;   // max 159
            amat2[j][e] = w2h[K];
        }
    }

    const float* Yimg = Y + (size_t)img * FT;

    for (int tc = 0; tc < TN; tc += TC) {
        if (tc + TC < TN) __builtin_prefetch(Yimg + f0 * TN + tc + TC, 0, 0);
        __syncthreads();   // protect xsf/h1 reuse across chunks

        // stage x tile: async global->LDS copy for in-range cells (ASYNCcnt path),
        // plain ds_store zeros for the padded halo (disjoint addresses).
        for (int i = tid; i < XR * XW; i += 256) {
            int r = i / XW, cidx = i % XW;
            int gf = f0 - 2 + r, gt = tc - 4 + cidx;
            if (gf >= 0 && gf < FQ && gt >= 0 && gt < TN) {
                unsigned lds_off = (unsigned)(uintptr_t)(&xsf[i]);
                unsigned long long gaddr =
                    (unsigned long long)(uintptr_t)(Yimg + gf * TN + gt);
                asm volatile("global_load_async_to_lds_b32 %0, %1, off"
                             :: "v"(lds_off), "v"(gaddr) : "memory");
            } else {
                xsf[i] = 0.0f;
            }
        }
        asm volatile("s_wait_asynccnt 0" ::: "memory");
        __syncthreads();

        // conv1: 18 rows x 5 tiles of 16 pixels -> 90 WMMA ops, split across 8 waves
        for (int w = wv; w < H1R * 5; w += 8) {
            int fr = w / 5, ti = w % 5;
            int t0 = ti * 16;
            int n  = lane & 15;
            v16h bmat;
            for (int e = 0; e < 16; ++e) bmat[e] = (_Float16)0.0f;
            if (lane < 16) {   // B 32x16: lanes 0-15 col N=lane hold K0..15; K>=9 zero
                for (int df = 0; df < 3; ++df)
                    for (int dt = 0; dt < 3; ++dt)
                        bmat[df * 3 + dt] =
                            (_Float16)xsf[(fr + df) * XW + t0 + n + dt];
            }
            v8f d = {};
            d = __builtin_amdgcn_wmma_f32_16x16x32_f16(
                    false, amat, false, bmat, (short)0, d, false, false);
            // D: lanes 0-15 hold channels 0..7 of pixel n; lanes 16-31 channels 8..15
            int gt = tc - 2 + t0 + n;
            int gf = f0 - 1 + fr;
            bool inb = (gt >= 0) && (gt < TN) && (gf >= 0) && (gf < FQ);
            int cbase = (lane < 16) ? 0 : 8;
            for (int r = 0; r < 8; ++r) {
                float hv = d[r] + b1s[cbase + r];
                hv = hv > 0.0f ? hv : 0.0f;     // ReLU
                if (!inb) hv = 0.0f;            // conv2's zero padding of h1
                h1[cbase + r][fr][t0 + n] = (_Float16)hv;
            }
        }
        __syncthreads();

        // conv2 via WMMA: 16 f rows x 4 t-tiles of 16 pixels, K=144 in 5 chunks
        for (int w = wv; w < 64; w += 8) {
            int floc = w >> 2, ti = w & 3;
            int tl0 = ti * 16;
            int n = lane & 15;
            v8f d2 = {};
            for (int j = 0; j < 5; ++j) {
                v16h bm;
                for (int e = 0; e < 16; ++e) {
                    int khalf = (e < 8) ? e : (e - 8 + 16);
                    int K = 32 * j + ((lane >= 16) ? 8 : 0) + khalf;
                    _Float16 val = (_Float16)0.0f;
                    if (K < 144) {
                        int c = K / 9, tap = K % 9;
                        int df = tap / 3, dt = tap % 3;
                        val = h1[c][floc + df][tl0 + n + dt];
                    }
                    bm[e] = val;
                }
                d2 = __builtin_amdgcn_wmma_f32_16x16x32_f16(
                        false, amat2[j], false, bm, (short)0, d2, false, false);
            }
            // every D row equals the conv2 output; take row 0 (lanes 0-15, vgpr 0)
            if (lane < 16)
                h2[((size_t)img * FQ + f0 + floc) * TN + tc + tl0 + n] = d2[0] + b2v;
        }
    }
}

// ---------------- softmax over t (512) per (img,f) row, + mask floor ----------------
__global__ __launch_bounds__(256) void softmax_kernel(const float* __restrict__ h2,
                                                      float* __restrict__ masks) {
    int row  = blockIdx.x * 8 + (threadIdx.x >> 5);   // 2048 rows, wave per row
    int lane = threadIdx.x & 31;
    const float* src = h2 + (size_t)row * TN;
    float vals[16];
    float m = -3.4e38f;
    for (int k = 0; k < 16; ++k) {
        vals[k] = src[lane + k * 32];
        m = fmaxf(m, vals[k]);
    }
    for (int d = 16; d >= 1; d >>= 1) m = fmaxf(m, __shfl_xor(m, d, 32));
    float s = 0.0f;
    for (int k = 0; k < 16; ++k) { vals[k] = __expf(vals[k] - m); s += vals[k]; }
    for (int d = 16; d >= 1; d >>= 1) s += __shfl_xor(s, d, 32);
    float inv = (1.0f - 1e-5f) / s;
    float* dst = masks + (size_t)row * TN;
    for (int k = 0; k < 16; ++k) dst[lane + k * 32] = vals[k] * inv + 1e-5f;
}

// ---------------- one ISS rank-1 update for a given source ----------------
// 512 threads = 16 wave32; all 8 cumulative sums done with ONE packed block scan
// (per-wave shfl scans + 2 barriers total instead of 8 x 3).
__global__ __launch_bounds__(512) void iss_kernel(float* __restrict__ Y,
                                                  const float* __restrict__ masks,
                                                  float* __restrict__ W,
                                                  float* __restrict__ A,
                                                  const float* __restrict__ reg,
                                                  int src) {
    __shared__ float part[16][9];     // [wave][k], padded to 9 to spread banks
    int b = blockIdx.x >> 8;          // 256 f per b
    int f = blockIdx.x & 255;
    int t = threadIdx.x;              // one t per thread
    int lane = t & 31, wv = t >> 5;

    size_t yb = ((size_t)b * SG) * FT + (size_t)f * TN + t;   // Y[b][0][f][t]
    float rg = reg[f];
    float regsq = rg * rg;
    float Ys = Y[yb + (size_t)src * FT];

    float Yv[4], mv[4];
    for (int s = 0; s < 4; ++s) {
        Yv[s] = Y[yb + (size_t)s * FT];
        mv[s] = masks[yb + (size_t)s * FT];
    }

    float p[8];
    for (int s = 0; s < 4; ++s) {
        p[s]     = mv[s] * Yv[s] * Ys;   // -> num
        p[4 + s] = mv[s] * Ys * Ys;      // -> den
    }
    // per-wave inclusive scans of all 8 streams
    for (int k = 0; k < 8; ++k)
        for (int d = 1; d < 32; d <<= 1) {
            float n = __shfl_up(p[k], d, 32);
            if (lane >= d) p[k] += n;
        }
    if (lane == 31)
        for (int k = 0; k < 8; ++k) part[wv][k] = p[k];
    __syncthreads();
    if (wv < 8) {                     // wave k scans the 16 partials of stream k
        float w = (lane < 16) ? part[lane][wv] : 0.0f;
        for (int d = 1; d < 16; d <<= 1) {
            float n = __shfl_up(w, d, 32);
            if (lane >= d) w += n;
        }
        if (lane < 16) part[lane][wv] = w;
    }
    __syncthreads();
    if (wv > 0)
        for (int k = 0; k < 8; ++k) p[k] += part[wv - 1][k];

    float invn = 1.0f / (float)(t + 1);
    float v[4], den_src = p[4 + src] * invn;
    for (int s = 0; s < 4; ++s)
        v[s] = (p[s] * invn) / (regsq + p[4 + s] * invn);
    float dsv = sqrtf(regsq + den_src);
    v[src] = 1.0f - 1.0f / dsv;

    size_t wb = (((size_t)b * FQ + f) * TN + t) * 16;   // W/A[b,f,t, 4,4]
    float ws[4], Am[16];
    for (int c = 0; c < 4; ++c) ws[c] = W[wb + src * 4 + c];
    for (int i = 0; i < 16; ++i) Am[i] = A[wb + i];

    for (int s = 0; s < 4; ++s) Y[yb + (size_t)s * FT] = Yv[s] - v[s] * Ys;
    for (int s = 0; s < 4; ++s)
        for (int c = 0; c < 4; ++c)
            W[wb + s * 4 + c] -= v[s] * ws[c];
    for (int c = 0; c < 4; ++c) {
        float a = dsv * (Am[c * 4 + 0] * v[0] + Am[c * 4 + 1] * v[1] +
                         Am[c * 4 + 2] * v[2] + Am[c * 4 + 3] * v[3]);
        A[wb + c * 4 + src] = Am[c * 4 + src] + a;
    }
}

// ---------------- writeout Y (B,S,f,T) -> out (B,T,F) ----------------
__global__ __launch_bounds__(256) void writeout(const float* __restrict__ Y,
                                                float* __restrict__ out) {
    int idx = blockIdx.x * 256 + threadIdx.x;     // over B*T*F
    int b = idx / (TN * FN);
    int rem = idx % (TN * FN);
    int t = rem / FN, Fi = rem % FN;
    int s = Fi / FQ, f = Fi % FQ;
    out[idx] = Y[((size_t)(b * SG + s)) * FT + (size_t)f * TN + t];
}

extern "C" void kernel_launch(void* const* d_in, const int* in_sizes, int n_in,
                              void* d_out, int out_size, void* d_ws, size_t ws_size,
                              hipStream_t stream) {
    (void)in_sizes; (void)n_in; (void)out_size; (void)ws_size;
    const float* x   = (const float*)d_in[0];
    const float* reg = (const float*)d_in[1];
    const float* w1  = (const float*)d_in[2];
    const float* b1  = (const float*)d_in[3];
    const float* w2  = (const float*)d_in[4];
    const float* b2  = (const float*)d_in[5];

    // workspace layout (floats): Y | masks | h2 | W | A  => 11 * 2^20 floats = 44 MB
    float* Yb = (float*)d_ws;
    float* Mb = Yb + (1 << 20);
    float* Hb = Mb + (1 << 20);
    float* Wb = Hb + (1 << 20);
    float* Ab = Wb + (4 << 20);

    reshape_in<<<4096, 256, 0, stream>>>(x, Yb);
    init_WA<<<16384, 256, 0, stream>>>(Wb, Ab);

    for (int it = 0; it < NITER; ++it) {
        conv_kernel<<<128, 256, 0, stream>>>(Yb, w1, b1, w2, b2, Hb);
        softmax_kernel<<<256, 256, 0, stream>>>(Hb, Mb);
        for (int s = 0; s < SG; ++s)
            iss_kernel<<<512, 512, 0, stream>>>(Yb, Mb, Wb, Ab, reg, s);
    }

    writeout<<<4096, 256, 0, stream>>>(Yb, (float*)d_out);
}
